// MultiHeadedAttention_28552942584286
// MI455X (gfx1250) — matrix operand order repaired and checked by reading, not compiled, not run on hardware
//
#include <hip/hip_runtime.h>
#include <hip/hip_bf16.h>

// ---------------------------------------------------------------------------
// MultiHeadedAttention for gfx1250 (MI455X)
// All GEMMs via v_wmma_f32_16x16x32_bf16; K/V tile staging via Tensor Data
// Mover (tensor_load_to_lds + s_wait_tensorcnt), double-buffered so the TDM
// streams tile kt+1 while waves compute tile kt. Mask is pre-packed to bits.
// B=2, S=2048, D=1024, H=16, DK=64
// ---------------------------------------------------------------------------

constexpr int kB  = 2;
constexpr int kS  = 2048;
constexpr int kD  = 1024;
constexpr int kH  = 16;
constexpr int kDK = 64;
constexpr int kM  = kB * kS;       // 4096 rows for the dense GEMMs
constexpr int kSW = kS / 32;       // mask words per row (64)

typedef __attribute__((ext_vector_type(16))) __bf16 v16bf;
typedef __attribute__((ext_vector_type(8)))  float  v8f;
typedef unsigned u32x4 __attribute__((ext_vector_type(4)));
typedef int      i32x8 __attribute__((ext_vector_type(8)));
typedef int      i32x4 __attribute__((ext_vector_type(4)));

union AFrag {            // 16x32 bf16 A-fragment or 32x16 bf16 B-fragment: 32 bytes
    v16bf v;
    uint4 u[2];
};

static __device__ __forceinline__ v8f wmma_bf16(const v16bf& a, const v16bf& b, const v8f& c) {
    return __builtin_amdgcn_wmma_f32_16x16x32_bf16(
        /*neg_a=*/false, a, /*neg_b=*/false, b,
        /*c_mod=*/(short)0, c, /*reuse_a=*/false, /*reuse_b=*/false);
}

static __device__ __forceinline__ unsigned pack_bf16(float a, float b) {
    union { __bf16 h[2]; unsigned u; } p;
    p.h[0] = (__bf16)a; p.h[1] = (__bf16)b;
    return p.u;
}

#if __has_builtin(__builtin_amdgcn_tensor_load_to_lds)
#define ATTN_USE_TDM 1
#else
#define ATTN_USE_TDM 0
#endif

#if ATTN_USE_TDM
// Issue one TDM load of a 2D bf16 tile (tile_w x tile_h, contiguous along w)
// from global memory into LDS. Per CDNA5 ISA D# layout:
//   group0: [1:0]=count=1, [63:32]=lds_addr, [120:64]=global_addr, [127:126]=type(2)
//   group1: data_size=1(2B), tensor_dim0/1, tile_dim0/1, tensor_dim0_stride
static __device__ __forceinline__ void tdm_load_tile_2d(
        unsigned lds_addr, const void* gptr,
        unsigned tile_w, unsigned tile_h,
        unsigned tensor_w, unsigned tensor_h, unsigned row_stride) {
    unsigned long long ga = (unsigned long long)gptr;
    u32x4 g0;
    g0.x = 1u;                                                  // count=1
    g0.y = lds_addr;                                            // LDS byte address
    g0.z = (unsigned)(ga & 0xFFFFFFFFull);                      // global_addr[31:0]
    g0.w = (unsigned)((ga >> 32) & 0x01FFFFFFull) | (2u << 30); // addr[56:32] | type=2
    i32x8 g1;
    g1[0] = (int)(1u << 16);                                    // data_size=1 -> 2 bytes
    g1[1] = (int)((tensor_w & 0xFFFFu) << 16);                  // tensor_dim0[15:0]
    g1[2] = (int)(((tensor_w >> 16) & 0xFFFFu) | ((tensor_h & 0xFFFFu) << 16));
    g1[3] = (int)(((tensor_h >> 16) & 0xFFFFu) | ((tile_w & 0xFFFFu) << 16));
    g1[4] = (int)(tile_h & 0xFFFFu);                            // tile_dim1 (tile_dim2=0)
    g1[5] = (int)row_stride;                                    // dim0_stride[31:0]
    g1[6] = 0;
    g1[7] = 0;
    i32x4 z4; z4[0] = 0; z4[1] = 0; z4[2] = 0; z4[3] = 0;
#if __clang_major__ >= 23
    i32x8 z8; z8[0]=0; z8[1]=0; z8[2]=0; z8[3]=0; z8[4]=0; z8[5]=0; z8[6]=0; z8[7]=0;
    __builtin_amdgcn_tensor_load_to_lds(g0, g1, z4, z4, z8, 0);
#else
    __builtin_amdgcn_tensor_load_to_lds(g0, g1, z4, z4, 0);
#endif
}
#endif

// ---------------------------------------------------------------------------
// Pack int32 mask [B,S,S] into a bitmask (1 bit per element, 1 MB total).
// Cuts attention-phase mask traffic ~32x; the bitmask is L2-resident.
// ---------------------------------------------------------------------------
__global__ __launch_bounds__(256)
void mask_pack_kernel(const int* __restrict__ mask, unsigned* __restrict__ bits) {
    const size_t w = (size_t)blockIdx.x * blockDim.x + threadIdx.x;
    const size_t nwords = (size_t)kB * kS * kSW;
    if (w >= nwords) return;
    const int* src = mask + w * 32;
    unsigned r = 0;
    #pragma unroll
    for (int i = 0; i < 32; ++i) r |= (src[i] != 0 ? 1u : 0u) << i;
    bits[w] = r;
}

// ---------------------------------------------------------------------------
// Dense GEMM:  C[m,n] = sum_k A[m,k] * W[n,k] + bias[n]   (i.e. x @ W^T + b)
// MODE 0: A = fp32 [kM,kD], out = bf16 scattered to per-head layout [B,H,S,DK]
// MODE 1: A = bf16 [kM,kD], out = fp32 row-major [kM,kD]  (final projection)
// Block tile 128x64, BK=32, double-buffered LDS: global loads for step k+1
// are in flight while the WMMAs of step k execute.
// ---------------------------------------------------------------------------
template <int MODE>
__global__ __launch_bounds__(256)
void gemm_kernel(const void* __restrict__ Aptr,
                 const float* __restrict__ W,
                 const float* __restrict__ bias,
                 void* __restrict__ Outptr) {
    __shared__ __bf16 lA[2][128][32];
    __shared__ __bf16 lB[2][64][32];

    const int tid  = threadIdx.x;
    const int lane = tid & 31;
    const int wave = tid >> 5;
    const int wm   = wave & 3;      // 0..3  -> 32-row strip
    const int wn   = wave >> 2;     // 0..1  -> 32-col strip
    const int half = lane >> 4;     // 0/1
    const int l15  = lane & 15;

    const int rowBase = blockIdx.x * 128;
    const int colBase = blockIdx.y * 64;

    // staging assignments (fixed per thread)
    const int aIdx = tid * 16;             // 128x32 A tile, 16 elems/thread
    const int ar = aIdx >> 5, ac = aIdx & 31;
    const int wIdx = tid * 8;              // 64x32 W tile, 8 elems/thread
    const int wr = wIdx >> 5, wc = wIdx & 31;

    float4 fa[4];
    uint4  ua[2];
    float4 fw[2];

    auto gload = [&](int kk) {
        if constexpr (MODE == 0) {
            const float* src = (const float*)Aptr + (size_t)(rowBase + ar) * kD + kk + ac;
            #pragma unroll
            for (int e = 0; e < 4; ++e) fa[e] = ((const float4*)src)[e];
        } else {
            const __bf16* src = (const __bf16*)Aptr + (size_t)(rowBase + ar) * kD + kk + ac;
            ua[0] = ((const uint4*)src)[0];
            ua[1] = ((const uint4*)src)[1];
        }
        const float* wsrc = W + (size_t)(colBase + wr) * kD + kk + wc;
        fw[0] = ((const float4*)wsrc)[0];
        fw[1] = ((const float4*)wsrc)[1];
    };
    auto lstore = [&](int buf) {
        if constexpr (MODE == 0) {
            #pragma unroll
            for (int e = 0; e < 4; ++e) {
                *(unsigned*)&lA[buf][ar][ac + e * 4 + 0] = pack_bf16(fa[e].x, fa[e].y);
                *(unsigned*)&lA[buf][ar][ac + e * 4 + 2] = pack_bf16(fa[e].z, fa[e].w);
            }
        } else {
            *(uint4*)&lA[buf][ar][ac + 0] = ua[0];
            *(uint4*)&lA[buf][ar][ac + 8] = ua[1];
        }
        *(unsigned*)&lB[buf][wr][wc + 0] = pack_bf16(fw[0].x, fw[0].y);
        *(unsigned*)&lB[buf][wr][wc + 2] = pack_bf16(fw[0].z, fw[0].w);
        *(unsigned*)&lB[buf][wr][wc + 4] = pack_bf16(fw[1].x, fw[1].y);
        *(unsigned*)&lB[buf][wr][wc + 6] = pack_bf16(fw[1].z, fw[1].w);
    };

    const v8f vzero = {0.f, 0.f, 0.f, 0.f, 0.f, 0.f, 0.f, 0.f};
    v8f acc[2][2];
    acc[0][0] = vzero; acc[0][1] = vzero; acc[1][0] = vzero; acc[1][1] = vzero;

    gload(0);
    lstore(0);
    constexpr int NIT = kD / 32;
    for (int it = 0; it < NIT; ++it) {
        const int buf = it & 1;
        __syncthreads();                 // publish LDS[buf]
        if (it + 1 < NIT) gload((it + 1) * 32);   // overlap with WMMA below

        AFrag a[2], b[2];
        #pragma unroll
        for (int mi = 0; mi < 2; ++mi) {
            const int r = wm * 32 + mi * 16 + l15;
            a[mi].u[0] = *(const uint4*)&lA[buf][r][half * 8];
            a[mi].u[1] = *(const uint4*)&lA[buf][r][16 + half * 8];
        }
        #pragma unroll
        for (int ni = 0; ni < 2; ++ni) {
            const int c = wn * 32 + ni * 16 + l15;
            b[ni].u[0] = *(const uint4*)&lB[buf][c][half * 8];
            b[ni].u[1] = *(const uint4*)&lB[buf][c][16 + half * 8];
        }
        #pragma unroll
        for (int mi = 0; mi < 2; ++mi)
            #pragma unroll
            for (int ni = 0; ni < 2; ++ni)
                acc[mi][ni] = wmma_bf16(a[mi].v, b[ni].v, acc[mi][ni]);

        if (it + 1 < NIT) lstore(buf ^ 1);        // other buffer: no conflict
    }

    // ---- epilogue ----
    #pragma unroll
    for (int mi = 0; mi < 2; ++mi) {
        #pragma unroll
        for (int ni = 0; ni < 2; ++ni) {
            #pragma unroll
            for (int e = 0; e < 8; ++e) {
                const int grow = rowBase + wm * 32 + mi * 16 + e + 8 * half;
                const int gcol = colBase + wn * 32 + ni * 16 + l15;
                const float val = acc[mi][ni][e] + bias[gcol];
                if constexpr (MODE == 0) {
                    // scatter to [B, H, S, DK] bf16
                    const int bb = grow / kS, ss = grow % kS;
                    const int hh = gcol >> 6, dk = gcol & 63;
                    ((__bf16*)Outptr)[(((size_t)bb * kH + hh) * kS + ss) * kDK + dk] = (__bf16)val;
                } else {
                    ((float*)Outptr)[(size_t)grow * kD + gcol] = val;
                }
            }
        }
    }
}

// ---------------------------------------------------------------------------
// Flash-style attention with the reference's non-standard scoring:
//   scores = exp(Q·K^T) / 8, masked to -1e9, then softmax, then P·V.
// Block: 128 q-rows of one (b,h). 8 waves, wave w owns rows [w*16, w*16+16)
// over the FULL 64-wide k tile -> softmax state is wave-private.
// K/V tiles: double-buffered TDM pipeline (wave 0 issues tensor_load_to_lds
// for tile kt+1 immediately after tile kt's s_wait_tensorcnt).
// ---------------------------------------------------------------------------
__global__ __launch_bounds__(256)
void attn_kernel(const __bf16* __restrict__ Q,
                 const __bf16* __restrict__ Kmat,
                 const __bf16* __restrict__ V,
                 const unsigned* __restrict__ mbits,
                 __bf16* __restrict__ Out) {
    __shared__ __bf16 lK[2][64][64];     // [j][d]
    __shared__ __bf16 lV[2][64][64];     // [j][d]  raw V tile (TDM target)
    __shared__ __bf16 lVt[2][64][64];    // [d][j]  transposed V tile
    __shared__ __bf16 lP[8][16][64];     // wave-private P tiles

    const int tid  = threadIdx.x;
    const int lane = tid & 31;
    const int wave = tid >> 5;
    const int half = lane >> 4;
    const int l15  = lane & 15;

    const int qt = blockIdx.x;           // q tile (128 rows)
    const int h  = blockIdx.y;
    const int b  = blockIdx.z;

    const size_t headOff = ((size_t)b * kH + h) * kS * kDK;
    const __bf16* Qh = Q    + headOff;
    const __bf16* Kh = Kmat + headOff;
    const __bf16* Vh = V    + headOff;
    const unsigned* mrowBase = mbits + (size_t)b * kS * kSW;

    // ---- Q fragments for this wave's 16 rows (resident for whole loop) ----
    AFrag qa[2];
    {
        const __bf16* qrow = Qh + (size_t)(qt * 128 + wave * 16 + l15) * kDK;
        #pragma unroll
        for (int ks = 0; ks < 2; ++ks) {
            qa[ks].u[0] = *(const uint4*)(qrow + ks * 32 + half * 8);
            qa[ks].u[1] = *(const uint4*)(qrow + ks * 32 + 16 + half * 8);
        }
    }

    const v8f vzero = {0.f, 0.f, 0.f, 0.f, 0.f, 0.f, 0.f, 0.f};
    v8f o[4];
    o[0] = vzero; o[1] = vzero; o[2] = vzero; o[3] = vzero;
    float mrow[8], lrow[8];
    #pragma unroll
    for (int e = 0; e < 8; ++e) { mrow[e] = -1e30f; lrow[e] = 0.f; }

    constexpr int NT = kS / 64;

#if ATTN_USE_TDM
    auto issue_tile = [&](int kt_, int buf) {
        tdm_load_tile_2d((unsigned)(size_t)&lK[buf][0][0],
                         Kh + (size_t)kt_ * 64 * kDK,
                         /*tile_w=*/64, /*tile_h=*/64,
                         /*tensor_w=*/64, /*tensor_h=*/kS, /*stride=*/64);
        tdm_load_tile_2d((unsigned)(size_t)&lV[buf][0][0],
                         Vh + (size_t)kt_ * 64 * kDK,
                         64, 64, 64, kS, 64);
    };
    if (tid < 32) issue_tile(0, 0);      // prime the pipeline
#endif

    for (int kt = 0; kt < NT; ++kt) {
        const int cur = kt & 1;

#if ATTN_USE_TDM
        if (tid < 32) __builtin_amdgcn_s_wait_tensorcnt(0);
        __syncthreads();                 // buf[cur] complete; prev compute done
        if (tid < 32 && kt + 1 < NT) issue_tile(kt + 1, cur ^ 1);  // overlap
#else
        {   // fallback: cooperative 32B copies
            const int r = tid >> 2;            // 64 rows, 4 threads/row
            const int p = (tid & 3) * 16;      // 16 bf16 per thread
            const __bf16* ksrc = Kh + (size_t)(kt * 64 + r) * kDK + p;
            *(uint4*)&lK[cur][r][p + 0] = ((const uint4*)ksrc)[0];
            *(uint4*)&lK[cur][r][p + 8] = ((const uint4*)ksrc)[1];
            const __bf16* vsrc = Vh + (size_t)(kt * 64 + r) * kDK + p;
            *(uint4*)&lV[cur][r][p + 0] = ((const uint4*)vsrc)[0];
            *(uint4*)&lV[cur][r][p + 8] = ((const uint4*)vsrc)[1];
        }
        __syncthreads();
#endif
        // prefetch tile kt+2 into GL2 (TDM for kt+1 is already in flight)
        if (kt + 2 < NT) {
            const size_t nro = (size_t)((kt + 2) * 64 + (tid >> 2)) * kDK + (tid & 3) * 16;
            __builtin_prefetch(Kh + nro, 0, 1);
            __builtin_prefetch(Vh + nro, 0, 1);
        }

        // ---- LDS->LDS transpose: lVt[cur][d][j] = lV[cur][j][d] ----
        {
            const int j  = tid & 63;
            const int d0 = (tid >> 6) * 16;
            uint4 c0 = *(const uint4*)&lV[cur][j][d0];
            uint4 c1 = *(const uint4*)&lV[cur][j][d0 + 8];
            const __bf16* e0 = (const __bf16*)&c0;
            const __bf16* e1 = (const __bf16*)&c1;
            #pragma unroll
            for (int i = 0; i < 8; ++i) lVt[cur][d0 + i][j]     = e0[i];
            #pragma unroll
            for (int i = 0; i < 8; ++i) lVt[cur][d0 + 8 + i][j] = e1[i];
        }
        __syncthreads();                 // lVt[cur] ready

        // ---- S = Q K^T  (16 rows x 64 cols per wave) ----
        v8f st[4];
        st[0] = vzero; st[1] = vzero; st[2] = vzero; st[3] = vzero;
        #pragma unroll
        for (int ks = 0; ks < 2; ++ks) {
            #pragma unroll
            for (int ni = 0; ni < 4; ++ni) {
                AFrag kb;
                const int c = ni * 16 + l15;   // k column j
                kb.u[0] = *(const uint4*)&lK[cur][c][ks * 32 + half * 8];
                kb.u[1] = *(const uint4*)&lK[cur][c][ks * 32 + 16 + half * 8];
                st[ni] = wmma_bf16(qa[ks].v, kb.v, st[ni]);
            }
        }

        // ---- score transform + bit-mask + online softmax (wave-private) ----
        const int kw = (kt * 64) >> 5;       // word offset of this k tile
        #pragma unroll
        for (int e = 0; e < 8; ++e) {
            const int qrow = qt * 128 + wave * 16 + e + 8 * half;
            const unsigned mw0 = mrowBase[(size_t)qrow * kSW + kw];
            const unsigned mw1 = mrowBase[(size_t)qrow * kSW + kw + 1];
            float rmax = -1e30f;
            #pragma unroll
            for (int ni = 0; ni < 4; ++ni) {
                const int kl = ni * 16 + l15;                 // 0..63 in tile
                const unsigned w = (ni < 2) ? mw0 : mw1;
                const bool keep = (w >> (kl & 31)) & 1u;
                float v = __expf(st[ni][e]) * 0.125f;
                v = keep ? v : -1e9f;
                st[ni][e] = v;
                rmax = fmaxf(rmax, v);
            }
            #pragma unroll
            for (int off = 1; off < 16; off <<= 1)
                rmax = fmaxf(rmax, __shfl_xor(rmax, off, 32));
            const float mnew  = fmaxf(mrow[e], rmax);
            const float scale = __expf(mrow[e] - mnew);
            float psum = 0.f;
            #pragma unroll
            for (int ni = 0; ni < 4; ++ni) {
                const float p = __expf(st[ni][e] - mnew);
                psum += p;
                lP[wave][e + 8 * half][ni * 16 + l15] = (__bf16)p;
            }
            #pragma unroll
            for (int off = 1; off < 16; off <<= 1)
                psum += __shfl_xor(psum, off, 32);
            lrow[e] = lrow[e] * scale + psum;
            mrow[e] = mnew;
            #pragma unroll
            for (int ni = 0; ni < 4; ++ni) o[ni][e] *= scale;
        }

        // ---- O += P V  (LDS is in-order per wave; lP is wave-private) ----
        #pragma unroll
        for (int ks = 0; ks < 2; ++ks) {
            AFrag pa;
            pa.u[0] = *(const uint4*)&lP[wave][l15][ks * 32 + half * 8];
            pa.u[1] = *(const uint4*)&lP[wave][l15][ks * 32 + 16 + half * 8];
            #pragma unroll
            for (int ni = 0; ni < 4; ++ni) {
                AFrag vb;
                const int c = ni * 16 + l15;   // output column d
                vb.u[0] = *(const uint4*)&lVt[cur][c][ks * 32 + half * 8];
                vb.u[1] = *(const uint4*)&lVt[cur][c][ks * 32 + 16 + half * 8];
                o[ni] = wmma_bf16(pa.v, vb.v, o[ni]);
            }
        }
    }

    // ---- normalize + merge heads: Out[b, s, h*64+d] (bf16) ----
    #pragma unroll
    for (int ni = 0; ni < 4; ++ni) {
        #pragma unroll
        for (int e = 0; e < 8; ++e) {
            const int qrow = qt * 128 + wave * 16 + e + 8 * half;
            const int dcol = ni * 16 + l15;
            const float val = o[ni][e] / lrow[e];
            Out[((size_t)b * kS + qrow) * kD + h * kDK + dcol] = (__bf16)val;
        }
    }
}

// ---------------------------------------------------------------------------
extern "C" void kernel_launch(void* const* d_in, const int* in_sizes, int n_in,
                              void* d_out, int out_size, void* d_ws, size_t ws_size,
                              hipStream_t stream) {
    (void)in_sizes; (void)n_in; (void)out_size; (void)ws_size;
    const float* query = (const float*)d_in[0];
    const float* key_  = (const float*)d_in[1];
    const float* value = (const float*)d_in[2];
    const int*   mask  = (const int*)  d_in[3];
    const float* Wq = (const float*)d_in[4];
    const float* bq = (const float*)d_in[5];
    const float* Wk = (const float*)d_in[6];
    const float* bk = (const float*)d_in[7];
    const float* Wv = (const float*)d_in[8];
    const float* bv = (const float*)d_in[9];
    const float* Wo = (const float*)d_in[10];
    const float* bo = (const float*)d_in[11];

    const size_t headElems = (size_t)kB * kH * kS * kDK;  // 4,194,304
    __bf16* Qws = (__bf16*)d_ws;
    __bf16* Kws = Qws + headElems;
    __bf16* Vws = Kws + headElems;
    __bf16* Aws = Vws + headElems;                        // [B*S, D] bf16
    unsigned* Mbits = (unsigned*)(Aws + headElems);       // 1 MB bitmask

    // pack mask to bits (independent of projections; runs first)
    {
        const size_t nwords = (size_t)kB * kS * kSW;      // 262,144
        const int blocks = (int)((nwords + 255) / 256);
        mask_pack_kernel<<<blocks, 256, 0, stream>>>(mask, Mbits);
    }

    dim3 gridG(kM / 128, kD / 64);   // 32 x 16
    gemm_kernel<0><<<gridG, 256, 0, stream>>>(query, Wq, bq, Qws);
    gemm_kernel<0><<<gridG, 256, 0, stream>>>(key_,  Wk, bk, Kws);
    gemm_kernel<0><<<gridG, 256, 0, stream>>>(value, Wv, bv, Vws);

    dim3 gridA(kS / 128, kH, kB);    // 16 x 16 x 2
    attn_kernel<<<gridA, 256, 0, stream>>>(Qws, Kws, Vws, Mbits, Aws);

    gemm_kernel<1><<<gridG, 256, 0, stream>>>(Aws, Wo, bo, (float*)d_out);
}
